// InfinityFormerAttention_60782377173517
// MI455X (gfx1250) — matrix operand order repaired and checked by reading, not compiled, not run on hardware
//
#include <hip/hip_runtime.h>

// ---------------- problem constants (from reference) ----------------
constexpr int Bc   = 2;
constexpr int Sc   = 2048;
constexpr int Hc   = 16;
constexpr int KVc  = 8;
constexpr int Dc   = 128;
constexpr int HIDc = 2048;
constexpr int Mc   = Bc * Sc;          // 4096 rows
constexpr int KVD  = KVc * Dc;         // 1024

typedef __bf16 bf16;
typedef __attribute__((ext_vector_type(16))) __bf16 v16bf;
typedef __attribute__((ext_vector_type(8)))  __bf16 v8bf;
typedef __attribute__((ext_vector_type(8)))  float  v8f;
typedef __attribute__((ext_vector_type(4)))  float  v4f;
typedef __attribute__((ext_vector_type(4)))  unsigned int u32x4;
typedef __attribute__((ext_vector_type(8)))  int i32x8;
typedef __attribute__((ext_vector_type(4)))  int i32x4;

union Frag16 { v16bf v; v8bf h[2]; };

__device__ __forceinline__ v8f zero8() {
  v8f z = {0.f,0.f,0.f,0.f,0.f,0.f,0.f,0.f};
  return z;
}

// CDNA5 async copy: 16B global -> LDS, tracked by ASYNCcnt.
__device__ __forceinline__ void async_copy16(const void* lds_ptr, const void* gptr) {
  unsigned lds_off = (unsigned)(size_t)lds_ptr;
  asm volatile("global_load_async_to_lds_b128 %0, %1, off"
               :: "v"(lds_off), "v"(gptr) : "memory");
}
#define S_WAIT_ASYNC(n) asm volatile("s_wait_asynccnt " #n ::: "memory")

// CDNA5 TDM: 2D tile load (bf16 elements) global -> LDS, tracked by TENSORcnt.
// D# group0: count=1, lds_addr, global_addr, type=2("image").
// D# group1: data_size=2B, tensor_dim0/1, tile_dim0/1, tensor_dim0_stride.
// Toolchain is the 6-arg builtin form: (u32x4, i32x8, i32x4, i32x4, i32x8, cpol).
__device__ __forceinline__ void tdm_load_2d(const void* lds_ptr, const void* gbase,
                                            unsigned dim0, unsigned dim1,
                                            unsigned stride0,
                                            unsigned tile0, unsigned tile1) {
  const unsigned lds_off = (unsigned)(size_t)lds_ptr;
  const size_t ga = (size_t)gbase;
  u32x4 g0;
  g0[0] = 1u;                                          // count=1 (valid, user)
  g0[1] = lds_off;                                     // lds_addr
  g0[2] = (unsigned)(ga & 0xffffffffu);                // global_addr[31:0]
  g0[3] = (unsigned)((ga >> 32) & 0x1ffffffu) | (2u << 30);  // ga[56:32], type=2
  i32x8 g1;
  g1[0] = (int)(1u << 16);                             // data_size=1 (2 bytes)
  g1[1] = (int)((dim0 & 0xffffu) << 16);               // tensor_dim0[15:0]
  g1[2] = (int)(((dim0 >> 16) & 0xffffu) | ((dim1 & 0xffffu) << 16));
  g1[3] = (int)(((dim1 >> 16) & 0xffffu) | ((tile0 & 0xffffu) << 16));
  g1[4] = (int)(tile1 & 0xffffu);                      // tile_dim1 (tile_dim2=0)
  g1[5] = (int)stride0;                                // tensor_dim0_stride[31:0]
  g1[6] = 0;
  g1[7] = 0;
  i32x4 z4 = {0, 0, 0, 0};
  i32x8 z8 = {0, 0, 0, 0, 0, 0, 0, 0};
#if __has_builtin(__builtin_amdgcn_tensor_load_to_lds)
  __builtin_amdgcn_tensor_load_to_lds(g0, g1, z4, z4, z8, 0);
#else
  asm volatile("tensor_load_to_lds %0, %1, %2, %3"
               :: "s"(g0), "s"(g1), "s"(z4), "s"(z4) : "memory");
#endif
}
#if __has_builtin(__builtin_amdgcn_s_wait_tensorcnt)
#define WAIT_TENSOR(n) __builtin_amdgcn_s_wait_tensorcnt(n)
#else
#define WAIT_TENSOR(n) asm volatile("s_wait_tensorcnt " #n ::: "memory")
#endif

// =====================================================================
// GEMM:  Y[M,N] = X[M,K] * W[N,K]^T   (bf16 in via async-LDS, f32 acc/out)
// Block 128x128, 256 thr (8 waves), wave = 64x32 (4x2 WMMA tiles), K-step 32.
// Double-buffered GLOBAL_LOAD_ASYNC_TO_LDS pipeline.
// =====================================================================
#define GT_M 128
#define GT_N 128
#define GT_K 32
#define LDSTR 40   // bf16 elems per LDS row (80B, 16B multiple)

__global__ __launch_bounds__(256)
void gemm_xwT_wmma(const bf16* __restrict__ X, const bf16* __restrict__ W,
                   float* __restrict__ Y, int M, int N, int K) {
  __shared__ bf16 As[2][GT_M][LDSTR];
  __shared__ bf16 Bs[2][GT_N][LDSTR];

  const int tid  = threadIdx.x;
  const int lane = tid & 31;
  const int wave = tid >> 5;
  const int bm = blockIdx.y * GT_M;
  const int bn = blockIdx.x * GT_N;
  const int wm = (wave >> 2) * 64;
  const int wn = (wave & 3) * 32;
  const int l16  = lane & 15;
  const int half = lane >> 4;

  v8f acc[4][2];
#pragma unroll
  for (int i = 0; i < 4; ++i)
#pragma unroll
    for (int j = 0; j < 2; ++j) acc[i][j] = zero8();

  // async tile stage: 4 x 16B per thread per tile (2 for A, 2 for B)
  const int srow = tid >> 1;          // 0..127
  const int scb  = (tid & 1) * 2;     // chunk base 0 / 2  (chunks of 8 bf16)
  auto issue_tile = [&](int k0, int buf) {
    const bf16* xp = X + (size_t)(bm + srow) * K + k0;
    const bf16* wp = W + (size_t)(bn + srow) * K + k0;
#pragma unroll
    for (int i = 0; i < 2; ++i) {
      const int c = scb + i;
      async_copy16(&As[buf][srow][c * 8], xp + c * 8);
      async_copy16(&Bs[buf][srow][c * 8], wp + c * 8);
    }
  };

  const int ntiles = K / GT_K;
  issue_tile(0, 0);
  for (int t = 0; t < ntiles; ++t) {
    const int buf = t & 1;
    if (t + 1 < ntiles) {
      __syncthreads();                   // buf^1 consumers (iter t-1) done
      issue_tile((t + 1) * GT_K, buf ^ 1);
      if (t + 2 < ntiles)                // L2 warm-up for the tile after next
        __builtin_prefetch(X + (size_t)(bm + srow) * K + (t + 2) * GT_K, 0, 0);
      S_WAIT_ASYNC(0x4);                 // tile t's 4 ops done (in-order)
    } else {
      S_WAIT_ASYNC(0x0);
    }
    __syncthreads();

    Frag16 a[4], b[2];
#pragma unroll
    for (int i = 0; i < 4; ++i) {
      // A 16x32 bf16: lanes<16 row=l16 K{0..7,16..23}; lanes>=16 K{8..15,24..31}
      const bf16* row = &As[buf][wm + i * 16 + l16][half * 8];
      a[i].h[0] = *(const v8bf*)(row);
      a[i].h[1] = *(const v8bf*)(row + 16);
    }
#pragma unroll
    for (int j = 0; j < 2; ++j) {
      // B 32x16 bf16: lanes<16 col=l16 K{0..7,16..23}; lanes>=16 K{8..15,24..31}
      const bf16* col = &Bs[buf][wn + j * 16 + l16][half * 8];
      b[j].h[0] = *(const v8bf*)(col);
      b[j].h[1] = *(const v8bf*)(col + 16);
    }
#pragma unroll
    for (int i = 0; i < 4; ++i)
#pragma unroll
      for (int j = 0; j < 2; ++j)
        acc[i][j] = __builtin_amdgcn_wmma_f32_16x16x32_bf16(
            false, a[i].v, false, b[j].v, (short)0, acc[i][j], false, false);
  }

  const int half8 = half * 8;
#pragma unroll
  for (int i = 0; i < 4; ++i)
#pragma unroll
    for (int j = 0; j < 2; ++j)
#pragma unroll
      for (int r = 0; r < 8; ++r) {
        const int row = bm + wm + i * 16 + r + half8;
        const int col = bn + wn + j * 16 + l16;
        Y[(size_t)row * N + col] = acc[i][j][r];
      }
}

// =====================================================================
// RMSNorm(D=128) + RoPE + ReLU, one wave per (b,s,head); bf16 out with
// GQA head repeat.  rotate-half partner = lane^16.
// =====================================================================
__global__ __launch_bounds__(256)
void norm_rope_relu(const float* __restrict__ P,
                    const float* __restrict__ cosT, const float* __restrict__ sinT,
                    const float* __restrict__ nw,
                    bf16* __restrict__ Out, int NH, int rep) {
  const int gw   = (int)((blockIdx.x * blockDim.x + threadIdx.x) >> 5);
  const int lane = threadIdx.x & 31;
  const int h    = gw % NH;
  const int row  = gw / NH;
  if (row >= Mc) return;
  const int s  = row % Sc;
  const int d0 = lane * 4;

  v4f xv = *(const v4f*)(P + (size_t)row * (NH * Dc) + h * Dc + d0);
  float ss = xv[0]*xv[0] + xv[1]*xv[1] + xv[2]*xv[2] + xv[3]*xv[3];
#pragma unroll
  for (int m = 16; m >= 1; m >>= 1) ss += __shfl_xor(ss, m, 32);
  const float inv = rsqrtf(ss * (1.0f / 128.0f) + 1e-6f);

  v4f nv = *(const v4f*)(nw + d0);
  v4f cs = *(const v4f*)(cosT + (size_t)s * Dc + d0);
  v4f sn = *(const v4f*)(sinT + (size_t)s * Dc + d0);

  float xn[4], pn[4], o[4];
#pragma unroll
  for (int e = 0; e < 4; ++e) xn[e] = xv[e] * inv * nv[e];
#pragma unroll
  for (int e = 0; e < 4; ++e) pn[e] = __shfl_xor(xn[e], 16, 32);
#pragma unroll
  for (int e = 0; e < 4; ++e) {
    const float rot = (lane < 16) ? -pn[e] : pn[e];
    o[e] = fmaxf(xn[e] * cs[e] + rot * sn[e], 0.f);
  }
  for (int r = 0; r < rep; ++r) {
    bf16* op = Out + (size_t)row * (Hc * Dc) + (h * rep + r) * Dc + d0;
#pragma unroll
    for (int e = 0; e < 4; ++e) op[e] = (bf16)o[e];
  }
}

// flat fp32 -> bf16 cast
__global__ __launch_bounds__(256)
void cast_flat(const float* __restrict__ in, bf16* __restrict__ out, size_t n) {
  for (size_t i = blockIdx.x * (size_t)blockDim.x + threadIdx.x; i < n;
       i += (size_t)gridDim.x * blockDim.x)
    out[i] = (bf16)in[i];
}

// fp32 -> bf16 cast with GQA head repeat, [row, nh_in*D] -> [B,S,H,D]
__global__ __launch_bounds__(256)
void cast_rep(const float* __restrict__ in, bf16* __restrict__ out,
              int nh_in, int rep, size_t total) {
  for (size_t i = blockIdx.x * (size_t)blockDim.x + threadIdx.x; i < total;
       i += (size_t)gridDim.x * blockDim.x) {
    const size_t d   = i % Dc;
    const size_t h   = (i / Dc) % Hc;
    const size_t row = i / (Dc * Hc);
    out[i] = (bf16)in[(row * nh_in + h / (size_t)rep) * Dc + d];
  }
}

// fp32 -> bf16 cast + transpose to [B,H,D,S] (V for flash B-fragments)
__global__ __launch_bounds__(256)
void cast_rep_T(const float* __restrict__ in, bf16* __restrict__ outT,
                int nh_in, int rep, size_t total) {
  for (size_t i = blockIdx.x * (size_t)blockDim.x + threadIdx.x; i < total;
       i += (size_t)gridDim.x * blockDim.x) {
    const size_t d   = i % Dc;
    const size_t h   = (i / Dc) % Hc;
    const size_t row = i / (Dc * Hc);
    const size_t b   = row / Sc, s = row % Sc;
    const float v = in[(row * nh_in + h / (size_t)rep) * Dc + d];
    outT[((b * Hc + h) * Dc + d) * Sc + s] = (bf16)v;
  }
}

// =====================================================================
// Flash attention: bf16 WMMA, fp32 online softmax, TDM (tensor_load_to_lds)
// double-buffered K/V tiles.  Grid (S/64, H, B), 128 thr (4 waves x 16 rows).
// =====================================================================
#define AQ 64
#define AK 32

__global__ __launch_bounds__(128)
void flash_attn(const bf16* __restrict__ Q,    // [B,S,H,D]
                const bf16* __restrict__ Kb,   // [B,S,H,D]
                const bf16* __restrict__ VT,   // [B,H,D,S]
                float* __restrict__ O,         // [B,S,H,D]
                int causal) {
  __shared__ bf16 Ks[2][AK][Dc];         // [kpos][d]  (TDM rows contiguous)
  __shared__ bf16 Vt[2][Dc][AK];         // [d][kpos]
  __shared__ bf16 Ps[4][16][AK + 8];     // per-wave P tile

  const int tid  = threadIdx.x;
  const int lane = tid & 31;
  const int wave = tid >> 5;
  const int l16  = lane & 15;
  const int half = lane >> 4;
  const int q0   = blockIdx.x * AQ;
  const int h    = blockIdx.y;
  const int b    = blockIdx.z;
  const float scale = 0.08838834764831845f;   // 1/sqrt(128)

  // Q rows kept in registers as 4 A-fragments (d chunks of 32)
  Frag16 aq[4];
  {
    const bf16* qrow = Q + (((size_t)b * Sc + q0 + wave * 16 + l16) * Hc + h) * Dc;
#pragma unroll
    for (int c = 0; c < 4; ++c) {
      const int kb0 = c * 32 + half * 8;
      aq[c].h[0] = *(const v8bf*)(qrow + kb0);
      aq[c].h[1] = *(const v8bf*)(qrow + kb0 + 16);
    }
  }

  float mrow[8], lrow[8];
  v8f o[8];
#pragma unroll
  for (int i = 0; i < 8; ++i) { mrow[i] = -1e30f; lrow[i] = 0.f; }
#pragma unroll
  for (int t = 0; t < 8; ++t) o[t] = zero8();

  // TDM tile issue (wave 0 only; TDM ignores EXEC, one op per descriptor)
  auto issue_tile = [&](int kb, int buf) {
    // K tile: 32 rows (s) x 128 elems (d), row stride H*D
    tdm_load_2d(&Ks[buf][0][0],
                Kb + (((size_t)b * Sc + kb) * Hc + h) * Dc,
                /*dim0=*/Dc, /*dim1=*/Sc, /*stride0=*/Hc * Dc,
                /*tile0=*/Dc, /*tile1=*/AK);
    // V^T tile: 128 rows (d) x 32 elems (s), row stride S
    tdm_load_2d(&Vt[buf][0][0],
                VT + ((size_t)b * Hc + h) * Dc * Sc + kb,
                /*dim0=*/Sc, /*dim1=*/Dc, /*stride0=*/Sc,
                /*tile0=*/AK, /*tile1=*/Dc);
  };

  const int kend    = causal ? (q0 + AQ) : Sc;
  const int nblocks = kend / AK;
  if (wave == 0) issue_tile(0, 0);
  for (int t = 0; t < nblocks; ++t) {
    const int kb  = t * AK;
    const int buf = t & 1;
    if (t + 1 < nblocks) {
      __syncthreads();                 // buf^1 consumers (iter t-1) done
      if (wave == 0) {
        issue_tile((t + 1) * AK, buf ^ 1);
        WAIT_TENSOR(2);                // tile t's 2 TDM ops complete (in-order)
      }
    } else if (wave == 0) {
      WAIT_TENSOR(0);
    }
    __syncthreads();

    // S = Q * K^T
    v8f sacc[2];
    sacc[0] = zero8(); sacc[1] = zero8();
#pragma unroll
    for (int j = 0; j < 2; ++j)
#pragma unroll
      for (int c = 0; c < 4; ++c) {
        Frag16 bk;
        const bf16* kc = &Ks[buf][j * 16 + l16][c * 32 + half * 8];
        bk.h[0] = *(const v8bf*)(kc);
        bk.h[1] = *(const v8bf*)(kc + 16);
        sacc[j] = __builtin_amdgcn_wmma_f32_16x16x32_bf16(
            false, aq[c].v, false, bk.v, (short)0, sacc[j], false, false);
      }

    // online softmax + stage P (bf16) in LDS
#pragma unroll
    for (int i = 0; i < 8; ++i) {
      const int qr = q0 + wave * 16 + i + half * 8;
      float s0 = sacc[0][i] * scale;
      float s1 = sacc[1][i] * scale;
      if (causal) {
        if (kb + l16 > qr)      s0 = -1e30f;
        if (kb + 16 + l16 > qr) s1 = -1e30f;
      }
      float mx = fmaxf(s0, s1);
#pragma unroll
      for (int m = 8; m >= 1; m >>= 1) mx = fmaxf(mx, __shfl_xor(mx, m, 32));
      const float mnew = fmaxf(mrow[i], mx);
      const float corr = __expf(mrow[i] - mnew);
      const float p0 = __expf(s0 - mnew);
      const float p1 = __expf(s1 - mnew);
      float psum = p0 + p1;
#pragma unroll
      for (int m = 8; m >= 1; m >>= 1) psum += __shfl_xor(psum, m, 32);
      lrow[i] = lrow[i] * corr + psum;
      mrow[i] = mnew;
#pragma unroll
      for (int tt = 0; tt < 8; ++tt) o[tt][i] *= corr;
      Ps[wave][i + half * 8][l16]      = (bf16)p0;
      Ps[wave][i + half * 8][16 + l16] = (bf16)p1;
    }
    asm volatile("s_wait_dscnt 0x0" ::: "memory");  // same-wave LDS RAW

    Frag16 ap;
    const bf16* pr = &Ps[wave][l16][half * 8];
    ap.h[0] = *(const v8bf*)(pr);
    ap.h[1] = *(const v8bf*)(pr + 16);

    // O += P * V   (8 d-tiles of 16)
#pragma unroll
    for (int tt = 0; tt < 8; ++tt) {
      Frag16 bv;
      const bf16* vc = &Vt[buf][tt * 16 + l16][half * 8];
      bv.h[0] = *(const v8bf*)(vc);
      bv.h[1] = *(const v8bf*)(vc + 16);
      o[tt] = __builtin_amdgcn_wmma_f32_16x16x32_bf16(
          false, ap.v, false, bv.v, (short)0, o[tt], false, false);
    }
  }

  // epilogue: O / l
#pragma unroll
  for (int tt = 0; tt < 8; ++tt)
#pragma unroll
    for (int i = 0; i < 8; ++i) {
      const int row = q0 + wave * 16 + i + half * 8;
      const int d   = tt * 16 + l16;
      O[(((size_t)b * Sc + row) * Hc + h) * Dc + d] = o[tt][i] / lrow[i];
    }
}

// sigmoid-gate combine -> bf16 (feeds final WMMA GEMM)
__global__ __launch_bounds__(256)
void gate_combine(const float* __restrict__ gl, const float* __restrict__ ctx,
                  const float* __restrict__ memo, bf16* __restrict__ comb,
                  size_t n) {
  for (size_t i = blockIdx.x * (size_t)blockDim.x + threadIdx.x; i < n;
       i += (size_t)gridDim.x * blockDim.x) {
    const float g = 1.f / (1.f + __expf(-gl[i]));
    comb[i] = (bf16)(g * ctx[i] + (1.f - g) * memo[i]);
  }
}

// =====================================================================
extern "C" void kernel_launch(void* const* d_in, const int* in_sizes, int n_in,
                              void* d_out, int out_size, void* d_ws, size_t ws_size,
                              hipStream_t stream) {
  (void)in_sizes; (void)n_in; (void)out_size; (void)ws_size;
  const float* hidden = (const float*)d_in[0];
  const float* cosT   = (const float*)d_in[1];
  const float* sinT   = (const float*)d_in[2];
  const float* Wq     = (const float*)d_in[3];
  const float* Wk     = (const float*)d_in[4];
  const float* Wv     = (const float*)d_in[5];
  const float* Wo     = (const float*)d_in[6];
  const float* Wg     = (const float*)d_in[7];
  const float* qnw    = (const float*)d_in[8];
  const float* knw    = (const float*)d_in[9];
  const float* mem_k  = (const float*)d_in[10];
  const float* mem_v  = (const float*)d_in[11];
  float* out = (float*)d_out;

  char* ws = (char*)d_ws;
  size_t off = 0;
  auto alloc = [&](size_t bytes) {
    void* p = ws + off;
    off += (bytes + 255) & ~(size_t)255;
    return p;
  };
  const size_t TOT = (size_t)Mc * Hc * Dc;          // == Mc*HIDc
  bf16* hb   = (bf16*)alloc((size_t)Mc * HIDc * 2);
  bf16* wqb  = (bf16*)alloc((size_t)HIDc * HIDc * 2);
  bf16* wkb  = (bf16*)alloc((size_t)KVD * HIDc * 2);
  bf16* wvb  = (bf16*)alloc((size_t)KVD * HIDc * 2);
  bf16* wgb  = (bf16*)alloc((size_t)HIDc * HIDc * 2);
  bf16* wob  = (bf16*)alloc((size_t)HIDc * HIDc * 2);
  float* qproj = (float*)alloc((size_t)Mc * HIDc * 4);
  float* kproj = (float*)alloc((size_t)Mc * KVD * 4);
  float* vproj = (float*)alloc((size_t)Mc * KVD * 4);
  float* gproj = (float*)alloc((size_t)Mc * HIDc * 4);
  bf16*  qb   = (bf16*) alloc(TOT * 2);
  bf16*  kb   = (bf16*) alloc(TOT * 2);
  bf16*  vbT  = (bf16*) alloc(TOT * 2);
  bf16*  mkb  = (bf16*) alloc(TOT * 2);
  bf16*  mvT  = (bf16*) alloc(TOT * 2);
  float* ctx  = (float*)alloc(TOT * 4);
  float* memo = (float*)alloc(TOT * 4);
  bf16*  comb = (bf16*) alloc(TOT * 2);

  // 0) one-shot bf16 conversions
  cast_flat<<<4096, 256, 0, stream>>>(hidden, hb,  (size_t)Mc * HIDc);
  cast_flat<<<4096, 256, 0, stream>>>(Wq,     wqb, (size_t)HIDc * HIDc);
  cast_flat<<<4096, 256, 0, stream>>>(Wk,     wkb, (size_t)KVD * HIDc);
  cast_flat<<<4096, 256, 0, stream>>>(Wv,     wvb, (size_t)KVD * HIDc);
  cast_flat<<<4096, 256, 0, stream>>>(Wg,     wgb, (size_t)HIDc * HIDc);
  cast_flat<<<4096, 256, 0, stream>>>(Wo,     wob, (size_t)HIDc * HIDc);

  // 1) projections (async-pipelined WMMA GEMMs)
  gemm_xwT_wmma<<<dim3(HIDc/GT_N, Mc/GT_M), 256, 0, stream>>>(hb, wqb, qproj, Mc, HIDc, HIDc);
  gemm_xwT_wmma<<<dim3(KVD /GT_N, Mc/GT_M), 256, 0, stream>>>(hb, wkb, kproj, Mc, KVD,  HIDc);
  gemm_xwT_wmma<<<dim3(KVD /GT_N, Mc/GT_M), 256, 0, stream>>>(hb, wvb, vproj, Mc, KVD,  HIDc);
  gemm_xwT_wmma<<<dim3(HIDc/GT_N, Mc/GT_M), 256, 0, stream>>>(hb, wgb, gproj, Mc, HIDc, HIDc);

  // 2) epilogues
  norm_rope_relu<<<(Mc * Hc)  / 8, 256, 0, stream>>>(qproj, cosT, sinT, qnw, qb, Hc, 1);
  norm_rope_relu<<<(Mc * KVc) / 8, 256, 0, stream>>>(kproj, cosT, sinT, knw, kb, KVc, Hc / KVc);
  cast_rep_T<<<4096, 256, 0, stream>>>(vproj, vbT, KVc, Hc / KVc, TOT);
  cast_rep  <<<4096, 256, 0, stream>>>(mem_k, mkb, Hc, 1, TOT);
  cast_rep_T<<<4096, 256, 0, stream>>>(mem_v, mvT, Hc, 1, TOT);

  // 3) attentions (TDM-pipelined): memory (unmasked) + context (causal)
  flash_attn<<<dim3(Sc/AQ, Hc, Bc), 128, 0, stream>>>(qb, mkb, mvT, memo, 0);
  flash_attn<<<dim3(Sc/AQ, Hc, Bc), 128, 0, stream>>>(qb, kb,  vbT, ctx,  1);

  // 4) gate combine (bf16) + output projection
  gate_combine<<<4096, 256, 0, stream>>>(gproj, ctx, memo, comb, TOT);
  gemm_xwT_wmma<<<dim3(HIDc/GT_N, Mc/GT_M), 256, 0, stream>>>(comb, wob, out, Mc, HIDc, HIDc);
}
